// Steamboat_43946105373325
// MI455X (gfx1250) — matrix operand order repaired and checked by reading, not compiled, not run on hardware
//
#include <hip/hip_runtime.h>
#include <hip/hip_bf16.h>
#include <hip/hip_fp16.h>

typedef __attribute__((ext_vector_type(16))) _Float16 v16h;
typedef __attribute__((ext_vector_type(8)))  _Float16 v8h;
typedef __attribute__((ext_vector_type(8)))  float    v8f;

// Problem constants (match reference)
constexpr int N_NODES = 100000;
constexpr int K_NBR   = 32;
constexpr int D_IN    = 512;
constexpr int H_HEADS = 32;
constexpr int NTILES  = N_NODES / 16;   // 6250 row tiles, exact
constexpr int NPAIRS  = NTILES / 2;     // 3125 tile pairs, exact

// Workspace layout (bytes, 256-aligned)
constexpr size_t OFF_QB   = 0;                         // 16 chunks * 2 ctile * 32 lanes * 16 halves
constexpr size_t OFF_KB   = 32768;
constexpr size_t OFF_VB   = 65536;                     // 32 ctile * 32 lanes * 16 halves
constexpr size_t OFF_EGO  = 98304;                     // H floats
constexpr size_t OFF_QEMB = 98560;                     // N*H f32
constexpr size_t OFF_KEMB = OFF_QEMB + (size_t)N_NODES * H_HEADS * 4;
constexpr size_t OFF_ATTN = OFF_KEMB + (size_t)N_NODES * H_HEADS * 4;  // N*H f16

__device__ __forceinline__ float nonneg(float w) {
    // elu(w) + 1 = (w > 0) ? w + 1 : exp(w)
    return w > 0.0f ? w + 1.0f : __expf(w);
}

// Load one fragment-ready B tile fragment: 32 contiguous bytes per lane.
__device__ __forceinline__ v16h ldB(const _Float16* __restrict__ B, int slot, int lane) {
    return *(const v16h*)(B + ((size_t)slot * 32 + lane) * 16);
}

// Build an A fragment (16x32 f16 tile) from a row of f32 x.
// Lanes 0-15: row=l, K {c..c+7, c+16..c+23}; lanes 16-31: row=l-16, +8.
__device__ __forceinline__ v16h ldA_f32(const float* __restrict__ xrow, int kc, int koff) {
    const float4 xa = *(const float4*)(xrow + kc + koff);
    const float4 xb = *(const float4*)(xrow + kc + koff + 4);
    const float4 xc = *(const float4*)(xrow + kc + koff + 16);
    const float4 xd = *(const float4*)(xrow + kc + koff + 20);
    v16h a;
    a[0]  = (_Float16)xa.x; a[1]  = (_Float16)xa.y;
    a[2]  = (_Float16)xa.z; a[3]  = (_Float16)xa.w;
    a[4]  = (_Float16)xb.x; a[5]  = (_Float16)xb.y;
    a[6]  = (_Float16)xb.z; a[7]  = (_Float16)xb.w;
    a[8]  = (_Float16)xc.x; a[9]  = (_Float16)xc.y;
    a[10] = (_Float16)xc.z; a[11] = (_Float16)xc.w;
    a[12] = (_Float16)xd.x; a[13] = (_Float16)xd.y;
    a[14] = (_Float16)xd.z; a[15] = (_Float16)xd.w;
    return a;
}

#define WMMA_F16(A, B, C) \
    __builtin_amdgcn_wmma_f32_16x16x32_f16(false, (A), false, (B), (short)0, (C), false, false)

// ---------------------------------------------------------------------------
// K1: nonneg() on weights, cast to f16, pre-swizzled into WMMA B-fragment
// layout so GEMM kernels do one contiguous 32B/lane load per fragment.
// B tile (32K x 16N), 16-bit: lane l(0-15) holds col n=l, K rows 0..7,16..23 (j=0..15);
// lane l+16 holds col n=l, K rows 8..15,24..31.
// ---------------------------------------------------------------------------
__global__ void prep_weights(const float* __restrict__ qw,   // (H,D)
                             const float* __restrict__ kw,   // (H,D)
                             const float* __restrict__ es,   // (1,H)
                             const float* __restrict__ vw,   // (D,H)
                             _Float16* __restrict__ qB,
                             _Float16* __restrict__ kB,
                             _Float16* __restrict__ vB,
                             float* __restrict__ ego) {
    int t = blockIdx.x * blockDim.x + threadIdx.x;
    if (t >= H_HEADS * D_IN) return;
    int h = t / D_IN;
    int d = t % D_IN;

    // q/k: B = W.T slice, K-dim = d, col = h. chunk = 32 K-rows.
    {
        int kk    = d & 31;
        int chunk = d >> 5;
        int lane  = (((kk >> 3) & 1) << 4) + (h & 15);
        int ct    = h >> 4;                 // two 16-col tiles for H=32
        int j     = (kk & 7) + ((kk >> 4) << 3);
        int idx   = (((chunk * 2 + ct) * 32) + lane) * 16 + j;
        qB[idx] = (_Float16)nonneg(qw[t]);
        kB[idx] = (_Float16)nonneg(kw[t]);
    }
    // v: B = v_w.T (H x D), K-dim = h (single chunk of 32), col = d.
    {
        int ct   = d >> 4;                  // 32 column tiles over D
        int lane = (((h >> 3) & 1) << 4) + (d & 15);
        int j    = (h & 7) + ((h >> 4) << 3);
        int idx  = ((ct * 32) + lane) * 16 + j;
        vB[idx] = (_Float16)nonneg(vw[(size_t)d * H_HEADS + h]);
    }
    if (t < H_HEADS) ego[t] = nonneg(es[t]);
}

// ---------------------------------------------------------------------------
// K2: q_emb / k_emb = x @ nonneg(W).T / D via v_wmma_f32_16x16x32_f16.
// TWO 16-row tiles per wave (halves L2 B traffic, 2 WMMAs per B fragment),
// with register double-buffering of B so each WMMA's s_wait covers loads
// issued a full iteration earlier.
// ---------------------------------------------------------------------------
__global__ void qk_gemm(const float* __restrict__ x,
                        const _Float16* __restrict__ qB,
                        const _Float16* __restrict__ kB,
                        float* __restrict__ q_emb,
                        float* __restrict__ k_emb) {
    const int wave = threadIdx.x >> 5;
    const int lane = threadIdx.x & 31;
    const int pair = blockIdx.x * (blockDim.x >> 5) + wave;
    if (pair >= NPAIRS) return;
    const int tile0 = pair * 2;

    const int row0 = tile0 * 16 + (lane & 15);
    const int koff = (lane >= 16) ? 8 : 0;
    const float* __restrict__ xrow0 = x + (size_t)row0 * D_IN;
    const float* __restrict__ xrow1 = xrow0 + (size_t)16 * D_IN;

    v8f q00 = {}, q01 = {}, k00 = {}, k01 = {};
    v8f q10 = {}, q11 = {}, k10 = {}, k11 = {};

    // Prime B double buffer with chunk 0 fragments.
    v16h bq0 = ldB(qB, 0, lane);
    v16h bq1 = ldB(qB, 1, lane);
    v16h bk0 = ldB(kB, 0, lane);
    v16h bk1 = ldB(kB, 1, lane);

#pragma unroll 4
    for (int chunk = 0; chunk < D_IN / 32; ++chunk) {
        const int kc = chunk * 32;
        const v16h a0 = ldA_f32(xrow0, kc, koff);
        const v16h a1 = ldA_f32(xrow1, kc, koff);

        // Prefetch next chunk's B fragments before consuming current ones.
        const int nc = (chunk + 1 < D_IN / 32) ? chunk + 1 : chunk;
        const v16h nbq0 = ldB(qB, nc * 2 + 0, lane);
        const v16h nbq1 = ldB(qB, nc * 2 + 1, lane);
        const v16h nbk0 = ldB(kB, nc * 2 + 0, lane);
        const v16h nbk1 = ldB(kB, nc * 2 + 1, lane);

        q00 = WMMA_F16(a0, bq0, q00);
        q01 = WMMA_F16(a0, bq1, q01);
        k00 = WMMA_F16(a0, bk0, k00);
        k01 = WMMA_F16(a0, bk1, k01);
        q10 = WMMA_F16(a1, bq0, q10);
        q11 = WMMA_F16(a1, bq1, q11);
        k10 = WMMA_F16(a1, bk0, k10);
        k11 = WMMA_F16(a1, bk1, k11);

        bq0 = nbq0; bq1 = nbq1; bk0 = nbk0; bk1 = nbk1;
    }

    // C/D layout: lane 0-15 -> n=lane, M=r; lane 16-31 -> n=lane-16, M=8+r.
    const float scale = 1.0f / (float)D_IN;
    const int n  = lane & 15;
    const int mb = (lane >= 16) ? 8 : 0;
#pragma unroll
    for (int r = 0; r < 8; ++r) {
        size_t o0 = (size_t)(tile0 * 16 + mb + r) * H_HEADS;
        size_t o1 = o0 + (size_t)16 * H_HEADS;
        q_emb[o0 + n]      = q00[r] * scale;
        q_emb[o0 + 16 + n] = q01[r] * scale;
        k_emb[o0 + n]      = k00[r] * scale;
        k_emb[o0 + 16 + n] = k01[r] * scale;
        q_emb[o1 + n]      = q10[r] * scale;
        q_emb[o1 + 16 + n] = q11[r] * scale;
        k_emb[o1 + n]      = k10[r] * scale;
        k_emb[o1 + 16 + n] = k11[r] * scale;
    }
}

// ---------------------------------------------------------------------------
// K3: neighbor gather-sum + ego term + row-normalize. One wave per node,
// lane = head. src indices broadcast via shuffle -> each gathered k_emb row
// is a fully coalesced 128B load (resident in 192MB L2: k_emb is 12.8MB).
// ---------------------------------------------------------------------------
__global__ void gather_norm(const int* __restrict__ src,
                            const float* __restrict__ q_emb,
                            const float* __restrict__ k_emb,
                            const float* __restrict__ ego,
                            _Float16* __restrict__ attn) {
    const int wave = threadIdx.x >> 5;
    const int lane = threadIdx.x & 31;
    const long i = (long)blockIdx.x * (blockDim.x >> 5) + wave;
    if (i >= N_NODES) return;

    const int sv = src[i * K_NBR + lane];
    float acc = 0.0f;
#pragma unroll
    for (int j = 0; j < K_NBR; ++j) {
        int sj = __shfl(sv, j, 32);
        acc += k_emb[(size_t)sj * H_HEADS + lane];
    }
    const float q = q_emb[(size_t)i * H_HEADS + lane];
    const float score = ego[lane] * q * q + q * acc * (1.0f / (float)K_NBR);

    float nrm = score;                       // wave32 tree reduction
#pragma unroll
    for (int off = 16; off >= 1; off >>= 1) nrm += __shfl_xor(nrm, off, 32);
    nrm += 1e-9f;

    attn[(size_t)i * H_HEADS + lane] = (_Float16)(score / nrm);
}

// ---------------------------------------------------------------------------
// K4: res = sum_attn (N x 32) @ nonneg(v_w).T (32 x 512). K=32 fits a single
// WMMA; TWO 16-row tiles per wave, double-buffered B over 32 column tiles.
// ---------------------------------------------------------------------------
__global__ void out_gemm(const _Float16* __restrict__ attn,
                         const _Float16* __restrict__ vB,
                         float* __restrict__ out) {
    const int wave = threadIdx.x >> 5;
    const int lane = threadIdx.x & 31;
    const int pair = blockIdx.x * (blockDim.x >> 5) + wave;
    if (pair >= NPAIRS) return;
    const int tile0 = pair * 2;

    const int row0 = tile0 * 16 + (lane & 15);
    const int koff = (lane >= 16) ? 8 : 0;

    v16h a0, a1;
    {
        const v8h lo0 = *(const v8h*)(attn + (size_t)row0 * H_HEADS + koff);
        const v8h hi0 = *(const v8h*)(attn + (size_t)row0 * H_HEADS + koff + 16);
        const v8h lo1 = *(const v8h*)(attn + (size_t)(row0 + 16) * H_HEADS + koff);
        const v8h hi1 = *(const v8h*)(attn + (size_t)(row0 + 16) * H_HEADS + koff + 16);
#pragma unroll
        for (int j = 0; j < 8; ++j) {
            a0[j] = lo0[j]; a0[8 + j] = hi0[j];
            a1[j] = lo1[j]; a1[8 + j] = hi1[j];
        }
    }

    const int n  = lane & 15;
    const int mb = (lane >= 16) ? 8 : 0;

    v16h b = ldB(vB, 0, lane);               // prime double buffer
#pragma unroll 4
    for (int ct = 0; ct < D_IN / 16; ++ct) {
        const int nct = (ct + 1 < D_IN / 16) ? ct + 1 : ct;
        const v16h nb = ldB(vB, nct, lane);

        v8f c0 = {}, c1 = {};
        c0 = WMMA_F16(a0, b, c0);
        c1 = WMMA_F16(a1, b, c1);

        float* o0 = out + (size_t)(tile0 * 16 + mb) * D_IN + ct * 16 + n;
        float* o1 = o0 + (size_t)16 * D_IN;
#pragma unroll
        for (int r = 0; r < 8; ++r) {
            o0[(size_t)r * D_IN] = c0[r];
            o1[(size_t)r * D_IN] = c1[r];
        }
        b = nb;
    }
}

// ---------------------------------------------------------------------------
extern "C" void kernel_launch(void* const* d_in, const int* in_sizes, int n_in,
                              void* d_out, int out_size, void* d_ws, size_t ws_size,
                              hipStream_t stream) {
    const int*   adj = (const int*)d_in[0];     // (2, N*K): row 0 = src
    const float* x   = (const float*)d_in[1];   // (N, D)
    const float* qw  = (const float*)d_in[2];   // (H, D)
    const float* kw  = (const float*)d_in[3];   // (H, D)
    const float* es  = (const float*)d_in[4];   // (1, H)
    const float* vw  = (const float*)d_in[5];   // (D, H)
    float* out = (float*)d_out;                 // (N, D)

    char* ws = (char*)d_ws;
    _Float16* qB   = (_Float16*)(ws + OFF_QB);
    _Float16* kB   = (_Float16*)(ws + OFF_KB);
    _Float16* vB   = (_Float16*)(ws + OFF_VB);
    float*    ego  = (float*)(ws + OFF_EGO);
    float*    qemb = (float*)(ws + OFF_QEMB);
    float*    kemb = (float*)(ws + OFF_KEMB);
    _Float16* attn = (_Float16*)(ws + OFF_ATTN);

    (void)in_sizes; (void)n_in; (void)out_size; (void)ws_size;

    // K1: weight prep (H*D = 16384 threads)
    prep_weights<<<(H_HEADS * D_IN + 255) / 256, 256, 0, stream>>>(
        qw, kw, es, vw, qB, kB, vB, ego);

    // K2: q/k GEMM — 8 waves/block, one 32-row tile pair per wave
    {
        int wavesPerBlock = 8;
        int blocks = (NPAIRS + wavesPerBlock - 1) / wavesPerBlock;
        qk_gemm<<<blocks, wavesPerBlock * 32, 0, stream>>>(x, qB, kB, qemb, kemb);
    }

    // K3: gather + normalize — one wave per node
    {
        int wavesPerBlock = 8;
        int blocks = (N_NODES + wavesPerBlock - 1) / wavesPerBlock;
        gather_norm<<<blocks, wavesPerBlock * 32, 0, stream>>>(adj, qemb, kemb, ego, attn);
    }

    // K4: output GEMM — one 32-row tile pair per wave
    {
        int wavesPerBlock = 8;
        int blocks = (NPAIRS + wavesPerBlock - 1) / wavesPerBlock;
        out_gemm<<<blocks, wavesPerBlock * 32, 0, stream>>>(attn, vB, out);
    }
}
